// graphCRFmodel_73366631350576
// MI455X (gfx1250) — matrix operand order repaired and checked by reading, not compiled, not run on hardware
//
#include <hip/hip_runtime.h>
#include <hip/hip_bf16.h>
#include <cstdint>

typedef __attribute__((ext_vector_type(16))) _Float16 v16h;
typedef __attribute__((ext_vector_type(8)))  _Float16 v8h;
typedef __attribute__((ext_vector_type(8)))  float    v8f;

#define OUT_LABEL 192
#define OUT_INT   64
#define LDIM      256           // L = OUT_LABEL + OUT_INT
#define SEQ       1024          // S
#define NTOK      1025          // n = S + 1
#define JPAD      1056          // NTOK padded to a multiple of 32 (zero-filled)
#define BATCH     32
#define NEGC      (-1e10f)
#define ROWTILES  65            // ceil(1025/16)

__device__ __forceinline__ float wave_max32(float v) {
#pragma unroll
  for (int o = 16; o > 0; o >>= 1) v = fmaxf(v, __shfl_xor(v, o, 32));
  return v;
}
__device__ __forceinline__ float wave_sum32(float v) {
#pragma unroll
  for (int o = 16; o > 0; o >>= 1) v += __shfl_xor(v, o, 32);
  return v;
}

// ---------------------------------------------------------------------------
// wT[nc*256 + k] = (f16) w[k*256 + nc]  — so stage-2 WMMA B-fragments of w are
// two contiguous 16B loads per lane (K runs along the fast axis).
// ---------------------------------------------------------------------------
__global__ void convert_w_kernel(const float* __restrict__ w,
                                 _Float16* __restrict__ wT) {
  int k  = blockIdx.x;       // 0..255
  int nc = threadIdx.x;      // 0..255
  wT[nc * 256 + k] = (_Float16)w[k * 256 + nc];
}

// ---------------------------------------------------------------------------
// Pt[b][l][j] = softmax(in_logits[b,j,:])[l] * tagmask  (f16, TRANSPOSED, j
// padded to JPAD with zeros).  One wave per j-row.  The transposed layout makes
// stage-1 WMMA B-fragments contiguous along K=j.
// ---------------------------------------------------------------------------
__global__ void softmax_init_kernel(const float* __restrict__ outlogits,
                                    const float* __restrict__ outlogits1,
                                    _Float16* __restrict__ Pt) {
  int wv = threadIdx.x >> 5;
  int lane = threadIdx.x & 31;
  int row = blockIdx.x * 8 + wv;
  if (row >= BATCH * JPAD) return;
  int b = row / JPAD, i = row % JPAD;
  _Float16* colbase = Pt + (size_t)b * LDIM * JPAD + i;  // + l*JPAD per slot

  if (i >= NTOK) {            // zero padding rows (exact-zero A there anyway)
#pragma unroll
    for (int k = 0; k < 8; ++k)
      colbase[(size_t)(lane + 32 * k) * JPAD] = (_Float16)0.f;
    return;
  }

  bool tok = (i < SEQ);
  const float* src = tok ? (outlogits + ((size_t)b * SEQ + i) * OUT_LABEL)
                         : (outlogits1 + (size_t)b * OUT_INT);
  int nact = tok ? OUT_LABEL : OUT_INT;

  float v[8];
  float mx = -3.4e38f;
#pragma unroll
  for (int k = 0; k < 8; ++k) {
    int la = lane + 32 * k;                 // index within the ACTIVE span
    v[k] = (la < nact) ? src[la] : -3.4e38f;
    mx = fmaxf(mx, v[k]);
  }
  mx = wave_max32(mx);
  float s = 0.f;
#pragma unroll
  for (int k = 0; k < 8; ++k) {
    int la = lane + 32 * k;
    v[k] = (la < nact) ? __expf(v[k] - mx) : 0.f;
    s += v[k];
  }
  s = wave_sum32(s);
  float inv = 1.f / s;

#pragma unroll
  for (int k = 0; k < 8; ++k) {
    int l = lane + 32 * k;                  // slot in the full 256-wide row
    float pv = 0.f;
    if (tok) {
      if (l < OUT_LABEL) pv = v[k] * inv;   // active span is [0,192)
    } else {
      int ka = k - 6;                       // l-192 = lane + 32*(k-6)
      if (ka >= 0 && l >= OUT_LABEL) pv = v[ka] * inv;  // active [192,256)
    }
    colbase[(size_t)l * JPAD] = (_Float16)pv;
  }
}

// ---------------------------------------------------------------------------
// One propagation step:
//   G = (A @ P) @ w + b + in_logits (+ mask*NEG)
//   final=0:  P_out = softmax(G) * tagmask   (f16, transposed layout)
//   final=1:  write output slices to d_out   (f32)
// Block = 256 threads (8 waves), owns 16 output rows x 256 cols of one batch.
// Stage 1 exploits the f32-exact band of A (exp(-0.1 d^2) underflows to +0.0f
// for d >= 33) plus the dense last row/column (forced d=1).  B-fragments come
// straight from the transposed, zero-padded Pt in global (L2-resident), so the
// stage-1 loop is barrier-free with EXEC all-ones throughout.  Loads are issued
// first so the branchless A-fragment exp/select chain hides their latency.
// ---------------------------------------------------------------------------
__global__ void __launch_bounds__(256) prop_kernel(
    const _Float16* __restrict__ Pin,    // transposed [B][256][JPAD]
    _Float16* __restrict__ Pout,         // transposed [B][256][JPAD]
    const float* __restrict__ outlogits,
    const float* __restrict__ outlogits1,
    const int* __restrict__ length,
    const _Float16* __restrict__ wT,
    const float* __restrict__ bias,
    float* __restrict__ out,
    int final_pass) {
  __shared__ _Float16 sT[16 * 264];    // stage-1 result tile (f16), stride 264
  __shared__ float    sG[16 * 256];    // stage-2 logits for cross-wave softmax

  const int tid = threadIdx.x;
  const int wv = tid >> 5, lane = tid & 31;
  const int h = lane >> 4, c = lane & 15;     // fragment half / column
  const int bb = blockIdx.x / ROWTILES;
  const int i0 = (blockIdx.x % ROWTILES) * 16;
  const int len_b = length[bb];

  v8f acc[2];
  acc[0] = (v8f){}; acc[1] = (v8f){};

  // ---- stage 1: T = A @ P over the exact-in-f32 band of A -----------------
  int lo, hi;
  if (i0 >= SEQ) { lo = 0; hi = 32; }                 // last row is dense (d=1)
  else {
    lo = (i0 - 64 < 0) ? 0 : (i0 - 64) >> 5;
    hi = (i0 + 79) >> 5; if (hi > 32) hi = 32;
  }
  const int last_extra = (hi < 32) ? 1 : 0;           // always add j=1024 chunk
  const _Float16* Ptb = Pin + (size_t)bb * LDIM * JPAD;
  const _Float16* col0 = Ptb + (size_t)(wv * 32 + c) * JPAD + 8 * h;
  const _Float16* col1 = Ptb + (size_t)(wv * 32 + 16 + c) * JPAD + 8 * h;

  const int irow = i0 + c;
  const bool irow_last = (irow == SEQ);

  for (int step = lo; step <= hi + last_extra; ++step) {
    const int ch = (step <= hi) ? step : 32;
    const int j0 = ch * 32;

    // issue both B-fragment tile loads first (2x16B contiguous per lane,
    // K = 8h+e and 16+8h+e along the fast axis of transposed P)
    v8h b0lo = *(const v8h*)(col0 + j0), b0hi = *(const v8h*)(col0 + j0 + 16);
    v8h b1lo = *(const v8h*)(col1 + j0), b1hi = *(const v8h*)(col1 + j0 + 16);

    // prefetch next chunk's B columns (global_prefetch_b8)
    if (step < hi + last_extra) {
      const int jn = (((step + 1) <= hi) ? (step + 1) : 32) * 32;
      __builtin_prefetch(col0 + jn, 0, 3);
      __builtin_prefetch(col1 + jn, 0, 3);
    }

    // A fragment in registers (16x32 f16, ISA layout: K = base(r)+8h+p),
    // fully branchless: unconditional exp + mask selects.  This long VALU
    // chain overlaps the outstanding B loads.
    v16h afrag;
#pragma unroll
    for (int e = 0; e < 16; ++e) {
      int r = e >> 1, p = e & 1;
      int kk = ((r < 4) ? (2 * r) : (16 + 2 * (r - 4))) + 8 * h + p;
      int j = j0 + kk;
      int di = irow - j; di = (di < 0) ? -di : di;
      bool special = irow_last != (j == SEQ);         // last row/col: d = 1
      float d2 = special ? 1.f : (float)(di * di);
      float ev = __expf(-0.1f * d2);                  // underflows to +0 off-band
      bool on = (j < NTOK) & (j != irow) & ((j < len_b) | (j == SEQ));
      afrag[e] = (_Float16)(on ? ev : 0.f);
    }

    v16h bf0, bf1;
#pragma unroll
    for (int e = 0; e < 8; ++e) {
      bf0[e] = b0lo[e]; bf0[e + 8] = b0hi[e];
      bf1[e] = b1lo[e]; bf1[e + 8] = b1hi[e];
    }
    acc[0] = __builtin_amdgcn_wmma_f32_16x16x32_f16(false, afrag, false, bf0,
                                                    (short)0, acc[0], false, false);
    acc[1] = __builtin_amdgcn_wmma_f32_16x16x32_f16(false, afrag, false, bf1,
                                                    (short)0, acc[1], false, false);
  }

  // spill T tile to LDS as f16 (C/D layout: M = r+8h, N = c)
#pragma unroll
  for (int r = 0; r < 8; ++r) {
    sT[(r + 8 * h) * 264 + wv * 32 + c]      = (_Float16)acc[0][r];
    sT[(r + 8 * h) * 264 + wv * 32 + 16 + c] = (_Float16)acc[1][r];
  }
  __syncthreads();

  // ---- stage 2: G = T @ w (K=256 in 8 chunks of 32) ------------------------
  acc[0] = (v8f){}; acc[1] = (v8f){};
  const int m = lane & 15;
#pragma unroll
  for (int kc = 0; kc < 8; ++kc) {
    const int k0 = kc * 32;
    v8h alo = *(const v8h*)&sT[m * 264 + k0 + 8 * h];
    v8h ahi = *(const v8h*)&sT[m * 264 + k0 + 16 + 8 * h];
    v16h af;
#pragma unroll
    for (int e = 0; e < 8; ++e) { af[e] = alo[e]; af[e + 8] = ahi[e]; }

    const _Float16* q0 = wT + (size_t)(wv * 32 + c) * 256 + k0 + 8 * h;
    const _Float16* q1 = wT + (size_t)(wv * 32 + 16 + c) * 256 + k0 + 8 * h;
    v8h w0lo = *(const v8h*)q0, w0hi = *(const v8h*)(q0 + 16);
    v8h w1lo = *(const v8h*)q1, w1hi = *(const v8h*)(q1 + 16);
    v16h wf0, wf1;
#pragma unroll
    for (int e = 0; e < 8; ++e) {
      wf0[e] = w0lo[e]; wf0[e + 8] = w0hi[e];
      wf1[e] = w1lo[e]; wf1[e + 8] = w1hi[e];
    }
    acc[0] = __builtin_amdgcn_wmma_f32_16x16x32_f16(false, af, false, wf0,
                                                    (short)0, acc[0], false, false);
    acc[1] = __builtin_amdgcn_wmma_f32_16x16x32_f16(false, af, false, wf1,
                                                    (short)0, acc[1], false, false);
  }

  // ---- epilogue: + bias + in_logits (+mask), then softmax or output -------
#pragma unroll
  for (int t = 0; t < 2; ++t) {
    const int l = wv * 32 + t * 16 + c;
    const float bl = bias[l];
#pragma unroll
    for (int r = 0; r < 8; ++r) {
      const int i = i0 + r + 8 * h;
      float inl;
      if (i < SEQ)
        inl = (l < OUT_LABEL) ? outlogits[((size_t)bb * SEQ + i) * OUT_LABEL + l]
                              : 2.f * NEGC;
      else
        inl = (l >= OUT_LABEL) ? outlogits1[(size_t)bb * OUT_INT + (l - OUT_LABEL)]
                               : 2.f * NEGC;
      const float val = acc[t][r] + bl + inl;
      if (final_pass) {
        if (i < SEQ) {
          if (l < OUT_LABEL)
            out[((size_t)bb * SEQ + i) * OUT_LABEL + l] = val;
        } else if (i == SEQ && l >= OUT_LABEL) {
          out[(size_t)BATCH * SEQ * OUT_LABEL + (size_t)bb * OUT_INT +
              (l - OUT_LABEL)] = val;
        }
      } else {
        sG[(r + 8 * h) * 256 + l] = val;
      }
    }
  }

  if (!final_pass) {
    __syncthreads();
#pragma unroll
    for (int rr = 0; rr < 2; ++rr) {
      const int mrow = wv * 2 + rr;
      const int i = i0 + mrow;
      if (i > SEQ) continue;                 // uniform per wave
      const bool tok = (i < SEQ);
      float v[8];
      float mx = -3.4e38f;
#pragma unroll
      for (int k = 0; k < 8; ++k) {
        int l = lane + 32 * k;
        bool act = tok ? (l < OUT_LABEL) : (l >= OUT_LABEL);
        v[k] = act ? sG[mrow * 256 + l] : -3.4e38f;
        mx = fmaxf(mx, v[k]);
      }
      mx = wave_max32(mx);
      float s = 0.f;
#pragma unroll
      for (int k = 0; k < 8; ++k) {
        int l = lane + 32 * k;
        bool act = tok ? (l < OUT_LABEL) : (l >= OUT_LABEL);
        v[k] = act ? __expf(v[k] - mx) : 0.f;
        s += v[k];
      }
      s = wave_sum32(s);
      const float inv = 1.f / s;
      _Float16* dstc = Pout + (size_t)bb * LDIM * JPAD + i;  // + l*JPAD
#pragma unroll
      for (int k = 0; k < 8; ++k) {
        int l = lane + 32 * k;
        dstc[(size_t)l * JPAD] = (_Float16)(v[k] * inv);
      }
    }
  }
}

// ---------------------------------------------------------------------------
extern "C" void kernel_launch(void* const* d_in, const int* in_sizes, int n_in,
                              void* d_out, int out_size, void* d_ws,
                              size_t ws_size, hipStream_t stream) {
  (void)in_sizes; (void)n_in; (void)out_size; (void)ws_size;
  const float* outlogits  = (const float*)d_in[0];  // [32,1024,192]
  const float* outlogits1 = (const float*)d_in[1];  // [32,64]
  const int*   length     = (const int*)d_in[2];    // [32]
  const float* w          = (const float*)d_in[3];  // [1,256,256]
  const float* bias       = (const float*)d_in[4];  // [256]
  float* out = (float*)d_out;

  char* ws = (char*)d_ws;
  const size_t Pbytes = (size_t)BATCH * LDIM * JPAD * sizeof(_Float16);
  _Float16* P0 = (_Float16*)ws;
  _Float16* P1 = (_Float16*)(ws + Pbytes);
  _Float16* wT = (_Float16*)(ws + 2 * Pbytes);

  convert_w_kernel<<<256, 256, 0, stream>>>(w, wT);
  softmax_init_kernel<<<(BATCH * JPAD) / 8, 256, 0, stream>>>(
      outlogits, outlogits1, P0);
  prop_kernel<<<BATCH * ROWTILES, 256, 0, stream>>>(
      P0, P1, outlogits, outlogits1, length, wT, bias, out, 0);
  prop_kernel<<<BATCH * ROWTILES, 256, 0, stream>>>(
      P1, nullptr, outlogits, outlogits1, length, wT, bias, out, 1);
}
